// SAGE_37563783971092
// MI455X (gfx1250) — matrix operand order repaired and checked by reading, not compiled, not run on hardware
//
#include <hip/hip_runtime.h>
#include <cstdint>

typedef __attribute__((ext_vector_type(16))) _Float16 v16h;
typedef __attribute__((ext_vector_type(8)))  _Float16 v8h;
typedef __attribute__((ext_vector_type(8)))  float    v8f;

static constexpr int cN1 = 262144, cN2 = 32768, cN3 = 8192;
static constexpr int cE0 = 1310720, cE1 = 327680, cE2 = 81920;
static constexpr int cF  = 100, cH = 256, cC = 47;

// ---------------------------------------------------------------------------
// Edge aggregation: one wave (32 lanes) per edge; coalesced row gather,
// global_atomic_add_f32 scatter into agg[dst], degree count.
// ---------------------------------------------------------------------------
template <typename T>
__global__ __launch_bounds__(256) void agg_kernel(
    const T* __restrict__ h, int ldh, int d,
    const int* __restrict__ esrc, const int* __restrict__ edst, int E,
    float* __restrict__ agg, int lda, float* __restrict__ deg)
{
    int gwave = (int)((blockIdx.x * blockDim.x + threadIdx.x) >> 5);
    int lane  = threadIdx.x & 31;
    if (gwave >= E) return;
    int src = esrc[gwave];
    int dst = edst[gwave];
    const T* hrow = h + (size_t)src * ldh;
    float*   arow = agg + (size_t)dst * lda;
    for (int c = lane; c < d; c += 32)
        atomicAdd(&arow[c], (float)hrow[c]);
    if (lane == 0)
        atomicAdd(&deg[dst], 1.0f);
}

// ---------------------------------------------------------------------------
// Prep passes (bandwidth-trivial) so GEMM staging is a pure byte copy.
// ---------------------------------------------------------------------------
// x [N1,100] f32 -> xh [N1,128] f16 zero-padded
__global__ __launch_bounds__(256) void pack_x_kernel(
    const float* __restrict__ x, _Float16* __restrict__ xh)
{
    size_t i = (size_t)blockIdx.x * 256 + threadIdx.x;   // over N1*128
    int m = (int)(i >> 7), k = (int)(i & 127);
    xh[i] = (k < cF) ? (_Float16)x[(size_t)m * cF + k] : (_Float16)0.0f;
}

// W [k_in, n_in] f32 row-major -> Wp [npad][kpad] f16, transposed+padded
__global__ __launch_bounds__(256) void pack_w_kernel(
    const float* __restrict__ W, int k_in, int n_in,
    _Float16* __restrict__ Wp, int kshift)
{
    size_t i = (size_t)blockIdx.x * 256 + threadIdx.x;   // over npad<<kshift
    int n = (int)(i >> kshift);
    int k = (int)(i & ((1 << kshift) - 1));
    Wp[i] = (n < n_in && k < k_in) ? (_Float16)W[(size_t)k * n_in + n]
                                   : (_Float16)0.0f;
}

// hn = (agg/deg) as f16 (mean-normalize), kpad = 1<<kshift per row
__global__ __launch_bounds__(256) void norm_kernel(
    const float* __restrict__ agg, const float* __restrict__ deg,
    _Float16* __restrict__ hn, int kshift)
{
    size_t i = (size_t)blockIdx.x * 256 + threadIdx.x;   // over M<<kshift
    size_t m = i >> kshift;
    float inv = __builtin_amdgcn_rcpf(fmaxf(deg[m], 1.0f));
    hn[i] = (_Float16)(agg[i] * inv);
}

// ---------------------------------------------------------------------------
// Fused SAGE layer GEMM:  out = [relu]( aself @ Ws + aneigh @ Wn + b )
// All operands pre-packed f16, K-contiguous. K iterated [self | neigh],
// 32 per step. Staging = GLOBAL_LOAD_ASYNC_TO_LDS_B128, double-buffered
// (issue tile s+1, s_wait_asynccnt 4 -> overlap DMA with 8-WMMA burst).
// Block: 256 threads (8 waves) -> 128(M) x 128(N); wave = 16x128 strip.
// ---------------------------------------------------------------------------
template <bool OUT_F16, bool RELU>
__global__ __launch_bounds__(256) void sage_gemm(
    const _Float16* __restrict__ aself,    // [M, kpad]
    const _Float16* __restrict__ aneigh,   // [M, kpad]
    const _Float16* __restrict__ wps,      // [npad, kpad] (N-major, K-contig)
    const _Float16* __restrict__ wpn,      // [npad, kpad]
    const float* __restrict__ bias,
    void* __restrict__ out_v, int ldo, int n_actual,
    int kpad)
{
    constexpr int ST   = 40;               // 32 + 8 halfs pad
    constexpr int BUFH = 128 * ST;         // halfs per buffer (10 KB)
    __shared__ _Float16 As[2 * BUFH];      // double-buffered A tiles
    __shared__ _Float16 Bs[2 * BUFH];      // double-buffered B tiles

    const int tid  = threadIdx.x;
    const int w    = tid >> 5;
    const int lane = tid & 31;
    const int hs   = lane >> 4;
    const int lr   = lane & 15;
    const int mBase = blockIdx.x * 128;
    const int nBase = blockIdx.y * 128;

    const int ksteps = kpad >> 5;          // steps per phase
    const int S      = 2 * ksteps;

    // per-thread staging assignment: rows r and r+64, 16B chunk c (fixed)
    const int srow = tid >> 2;
    const int sc   = (tid & 3) * 8;        // half offset of 16B chunk

    auto issue_tile = [&](int s) {
        const bool neigh = (s >= ksteps);
        const int  k0    = (neigh ? s - ksteps : s) * 32;
        const _Float16* ap = neigh ? aneigh : aself;
        const _Float16* wp = neigh ? wpn : wps;
        const int bufh = (s & 1) * BUFH;
#pragma unroll
        for (int i = 0; i < 2; ++i) {
            const int row = srow + i * 64;
            const _Float16* ga = ap + (size_t)(mBase + row) * kpad + k0 + sc;
            const _Float16* gb = wp + (size_t)(nBase + row) * kpad + k0 + sc;
            uint32_t la = (uint32_t)(uintptr_t)&As[bufh + row * ST + sc];
            uint32_t lb = (uint32_t)(uintptr_t)&Bs[bufh + row * ST + sc];
            asm volatile("global_load_async_to_lds_b128 %0, %1, off"
                         :: "v"(la), "v"((uint64_t)(uintptr_t)ga) : "memory");
            asm volatile("global_load_async_to_lds_b128 %0, %1, off"
                         :: "v"(lb), "v"((uint64_t)(uintptr_t)gb) : "memory");
        }
    };

    v8f acc[8] = {};

    issue_tile(0);
    for (int s = 0; s < S; ++s) {
        if (s + 1 < S) {
            issue_tile(s + 1);                                  // prefetch next
            asm volatile("s_wait_asynccnt 0x4" ::: "memory");   // tile s done
        } else {
            asm volatile("s_wait_asynccnt 0x0" ::: "memory");
        }
        __syncthreads();

        const int bufh = (s & 1) * BUFH;
        // A fragment: lane = row M=lr; K halves per ISA layout
        const int mrow = w * 16 + lr;
        v8h alo = *(const v8h*)&As[bufh + mrow * ST + hs * 8];
        v8h ahi = *(const v8h*)&As[bufh + mrow * ST + 16 + hs * 8];
        v16h a  = __builtin_shufflevector(alo, ahi,
                    0,1,2,3,4,5,6,7,8,9,10,11,12,13,14,15);

#pragma unroll
        for (int j = 0; j < 8; ++j) {
            // B fragment: lane = column N=lr; K halves per ISA layout (same as A)
            v8h blo = *(const v8h*)&Bs[bufh + (j * 16 + lr) * ST + hs * 8];
            v8h bhi = *(const v8h*)&Bs[bufh + (j * 16 + lr) * ST + 16 + hs * 8];
            v16h b  = __builtin_shufflevector(blo, bhi,
                        0,1,2,3,4,5,6,7,8,9,10,11,12,13,14,15);
            acc[j] = __builtin_amdgcn_wmma_f32_16x16x32_f16(
                         false, a, false, b, (short)0, acc[j], false, false);
        }
        __syncthreads();
    }

    // epilogue: bias (+relu); D layout: m = r + 8*hs, n = lane%16
#pragma unroll
    for (int j = 0; j < 8; ++j) {
        int n = nBase + j * 16 + lr;
        if (n < n_actual) {
            float bv = bias[n];
#pragma unroll
            for (int r = 0; r < 8; ++r) {
                size_t m = (size_t)(mBase + w * 16 + hs * 8 + r);
                float v = acc[j][r] + bv;
                if (RELU) v = fmaxf(v, 0.0f);
                if (OUT_F16)
                    ((_Float16*)out_v)[m * ldo + n] = (_Float16)v;
                else
                    ((float*)out_v)[m * ldo + n] = v;
            }
        }
    }
}

// ---------------------------------------------------------------------------
extern "C" void kernel_launch(void* const* d_in, const int* in_sizes, int n_in,
                              void* d_out, int out_size, void* d_ws, size_t ws_size,
                              hipStream_t stream)
{
    (void)in_sizes; (void)n_in; (void)out_size; (void)ws_size;

    const float* x   = (const float*)d_in[0];
    const int* es0   = (const int*)d_in[1];
    const int* ed0   = (const int*)d_in[2];
    const int* es1   = (const int*)d_in[3];
    const int* ed1   = (const int*)d_in[4];
    const int* es2   = (const int*)d_in[5];
    const int* ed2   = (const int*)d_in[6];
    const float* Ws0 = (const float*)d_in[7];
    const float* Wn0 = (const float*)d_in[8];
    const float* b0  = (const float*)d_in[9];
    const float* Ws1 = (const float*)d_in[10];
    const float* Wn1 = (const float*)d_in[11];
    const float* b1  = (const float*)d_in[12];
    const float* Ws2 = (const float*)d_in[13];
    const float* Wn2 = (const float*)d_in[14];
    const float* b2  = (const float*)d_in[15];

    char* ws = (char*)d_ws;
    size_t off = 0;
    auto wsalloc = [&](size_t bytes) -> void* {
        void* p = ws + off;
        off += (bytes + 255) & ~(size_t)255;
        return p;
    };
    float*    agg0 = (float*)   wsalloc((size_t)cN1 * 128 * 4);
    float*    deg0 = (float*)   wsalloc((size_t)cN1 * 4);
    _Float16* hn0  = (_Float16*)wsalloc((size_t)cN1 * 128 * 2);
    _Float16* xh   = (_Float16*)wsalloc((size_t)cN1 * 128 * 2);
    _Float16* h1   = (_Float16*)wsalloc((size_t)cN1 * cH * 2);
    float*    agg1 = (float*)   wsalloc((size_t)cN2 * cH * 4);
    float*    deg1 = (float*)   wsalloc((size_t)cN2 * 4);
    _Float16* hn1  = (_Float16*)wsalloc((size_t)cN2 * cH * 2);
    _Float16* h2   = (_Float16*)wsalloc((size_t)cN2 * cH * 2);
    float*    agg2 = (float*)   wsalloc((size_t)cN3 * cH * 4);
    float*    deg2 = (float*)   wsalloc((size_t)cN3 * 4);
    _Float16* hn2  = (_Float16*)wsalloc((size_t)cN3 * cH * 2);
    _Float16* Wp0s = (_Float16*)wsalloc((size_t)256 * 128 * 2);
    _Float16* Wp0n = (_Float16*)wsalloc((size_t)256 * 128 * 2);
    _Float16* Wp1s = (_Float16*)wsalloc((size_t)256 * 256 * 2);
    _Float16* Wp1n = (_Float16*)wsalloc((size_t)256 * 256 * 2);
    _Float16* Wp2s = (_Float16*)wsalloc((size_t)128 * 256 * 2);
    _Float16* Wp2n = (_Float16*)wsalloc((size_t)128 * 256 * 2);

    // Atomic accumulators must be re-zeroed every call (graph replay safe).
    hipMemsetAsync(agg0, 0, (size_t)cN1 * 128 * 4, stream);
    hipMemsetAsync(deg0, 0, (size_t)cN1 * 4, stream);
    hipMemsetAsync(agg1, 0, (size_t)cN2 * cH * 4, stream);
    hipMemsetAsync(deg1, 0, (size_t)cN2 * 4, stream);
    hipMemsetAsync(agg2, 0, (size_t)cN3 * cH * 4, stream);
    hipMemsetAsync(deg2, 0, (size_t)cN3 * 4, stream);

    // Weight pre-pack (tiny) + x pre-pack
    pack_w_kernel<<<(256 * 128) / 256, 256, 0, stream>>>(Ws0, cF, cH, Wp0s, 7);
    pack_w_kernel<<<(256 * 128) / 256, 256, 0, stream>>>(Wn0, cF, cH, Wp0n, 7);
    pack_w_kernel<<<(256 * 256) / 256, 256, 0, stream>>>(Ws1, cH, cH, Wp1s, 8);
    pack_w_kernel<<<(256 * 256) / 256, 256, 0, stream>>>(Wn1, cH, cH, Wp1n, 8);
    pack_w_kernel<<<(128 * 256) / 256, 256, 0, stream>>>(Ws2, cH, cC, Wp2s, 8);
    pack_w_kernel<<<(128 * 256) / 256, 256, 0, stream>>>(Wn2, cH, cC, Wp2n, 8);
    pack_x_kernel<<<((size_t)cN1 * 128) / 256, 256, 0, stream>>>(x, xh);

    // ---- Layer 0: x -> h1[262144,256] f16, relu ----
    agg_kernel<float><<<(cE0 + 7) / 8, 256, 0, stream>>>(
        x, cF, cF, es0, ed0, cE0, agg0, 128, deg0);
    norm_kernel<<<((size_t)cN1 * 128) / 256, 256, 0, stream>>>(agg0, deg0, hn0, 7);
    sage_gemm<true, true><<<dim3(cN1 / 128, 2), 256, 0, stream>>>(
        xh, hn0, Wp0s, Wp0n, b0, h1, cH, cH, 128);

    // ---- Layer 1: h1 -> h2[32768,256] f16, relu ----
    agg_kernel<_Float16><<<(cE1 + 7) / 8, 256, 0, stream>>>(
        h1, cH, cH, es1, ed1, cE1, agg1, cH, deg1);
    norm_kernel<<<((size_t)cN2 * cH) / 256, 256, 0, stream>>>(agg1, deg1, hn1, 8);
    sage_gemm<true, true><<<dim3(cN2 / 128, 2), 256, 0, stream>>>(
        h1, hn1, Wp1s, Wp1n, b1, h2, cH, cH, 256);

    // ---- Layer 2: h2 -> out[8192,47] f32, no relu ----
    agg_kernel<_Float16><<<(cE2 + 7) / 8, 256, 0, stream>>>(
        h2, cH, cH, es2, ed2, cE2, agg2, cH, deg2);
    norm_kernel<<<((size_t)cN3 * cH) / 256, 256, 0, stream>>>(agg2, deg2, hn2, 8);
    sage_gemm<false, false><<<dim3(cN3 / 128, 1), 256, 0, stream>>>(
        h2, hn2, Wp2s, Wp2n, b2, d_out, cC, cC, 256);
}
